// GAFN_Four_V2_19292993093866
// MI455X (gfx1250) — compile-verified, matrix operand order checked by reading, not run on hardware
//
#include <hip/hip_runtime.h>

// ---- CDNA5 / gfx1250: wave32, WMMA 16x16x32 f16 -> f32 ----
typedef __attribute__((ext_vector_type(16))) _Float16 v16h;
typedef __attribute__((ext_vector_type(8)))  float    v8f;
typedef __attribute__((ext_vector_type(4)))  float    v4f;

#define NB   8
#define CB   16
#define DB   49
#define FB   128
#define KB   64
#define RB   256
#define PIX  (CB*DB)        /* 784 pixels per batch */
#define TILE 16             /* pixels per workgroup (WMMA M) */
#define TPN  (PIX/TILE)     /* 49 tiles per batch  */
#define EPSV 1e-12f

__global__ __launch_bounds__(256)
void vlad_fused(const float* __restrict__ x,        // [N,C,D,F]
                const float* __restrict__ cluster,  // [N,R,F]
                const float* __restrict__ conv_w,   // [K,F]
                const float* __restrict__ conv_b,   // [K]
                const int*  __restrict__ row_idx,   // [K]
                float* __restrict__ out)            // [N,C,D,K*F]
{
  __shared__ float    s_xn[TILE][FB];     // normalized x, f32 (8 KB)
  __shared__ _Float16 s_xn16[TILE][FB];   // normalized x, f16 (4 KB)
  __shared__ _Float16 s_w16[KB][FB];      // conv_w, f16        (16 KB)
  __shared__ float    s_cent[KB][FB];     // gathered centroids (32 KB)
  __shared__ float    s_soft[TILE][KB];   // logits -> softmax  (4 KB)
  __shared__ float    s_nk[TILE][KB];     // per-(p,k) residual norms
  __shared__ float    s_scale[TILE][KB];  // final per-(p,k) scale

  const int tid  = threadIdx.x;
  const int lane = tid & 31;
  const int wave = tid >> 5;

  const int  tileId = blockIdx.x;
  const int  n   = tileId / TPN;
  const int  p0  = (tileId % TPN) * TILE;
  const long gp0 = (long)n * PIX + p0;    // global pixel base of this tile

  // ---------------- Stage 1: load & L2-normalize 16 rows of x ----------------
  {
    const int p  = tid >> 4;             // pixel 0..15 (16 lanes per pixel)
    const int fg = (tid & 15) * 8;       // 8 contiguous floats per lane
    const float* xp = x + (gp0 + p) * (long)FB + fg;
    v4f a = *(const v4f*)(xp);
    v4f b = *(const v4f*)(xp + 4);
    float ss = a.x*a.x + a.y*a.y + a.z*a.z + a.w*a.w
             + b.x*b.x + b.y*b.y + b.z*b.z + b.w*b.w;
    #pragma unroll
    for (int m = 1; m < 16; m <<= 1) ss += __shfl_xor(ss, m, 16);
    const float inv = 1.0f / fmaxf(sqrtf(ss), EPSV);
    v4f an = a * inv, bn = b * inv;
    *(v4f*)&s_xn[p][fg]     = an;
    *(v4f*)&s_xn[p][fg + 4] = bn;
    s_xn16[p][fg+0] = (_Float16)an.x;  s_xn16[p][fg+1] = (_Float16)an.y;
    s_xn16[p][fg+2] = (_Float16)an.z;  s_xn16[p][fg+3] = (_Float16)an.w;
    s_xn16[p][fg+4] = (_Float16)bn.x;  s_xn16[p][fg+5] = (_Float16)bn.y;
    s_xn16[p][fg+6] = (_Float16)bn.z;  s_xn16[p][fg+7] = (_Float16)bn.w;
  }

  // ---------------- Stage 2: stage conv_w (f16) and gathered centroids -------
  for (int i = tid; i < KB * FB / 4; i += 256) {     // 8 iterations
    const int k = i >> 5;
    const int f = (i & 31) * 4;
    v4f w = *(const v4f*)(conv_w + k * FB + f);
    s_w16[k][f+0] = (_Float16)w.x;  s_w16[k][f+1] = (_Float16)w.y;
    s_w16[k][f+2] = (_Float16)w.z;  s_w16[k][f+3] = (_Float16)w.w;
  }
  for (int k = wave; k < KB; k += 8) {               // 8 rows per wave
    const int row = row_idx[k];
    const float* cp = cluster + ((long)n * RB + row) * FB;
    *(v4f*)&s_cent[k][lane * 4] = *(const v4f*)(cp + lane * 4);
  }
  __syncthreads();

  // ---------------- Stage 3: logits GEMM via WMMA (waves 0..3) ---------------
  // M=16 pixels, N=64 clusters (one 16-wide N-tile per wave), K=128 (4x32).
  if (wave < 4) {
    const int m   = lane & 15;             // A row
    const int hi8 = (lane >> 4) * 8;       // K-subblock select (wave32 layout)
    const int nn  = (wave << 4) + m;       // global B column for this lane
    v8f acc = {};
    #pragma unroll
    for (int kk = 0; kk < 4; ++kk) {
      v16h a, b;
      #pragma unroll
      for (int v = 0; v < 8; ++v) {
        const int k = kk * 32 + ((v & 4) ? 16 : 0) + hi8 + 2 * (v & 3);
        a[2*v]   = s_xn16[m][k];
        a[2*v+1] = s_xn16[m][k + 1];
        b[2*v]   = s_w16[nn][k];
        b[2*v+1] = s_w16[nn][k + 1];
      }
      acc = __builtin_amdgcn_wmma_f32_16x16x32_f16(
                false, a, false, b, (short)0, acc, false, false);
    }
    const float bias = conv_b[nn];
    #pragma unroll
    for (int v = 0; v < 8; ++v) {
      const int M = v + hi8;               // D layout: VGPR v -> row v (+8 hi)
      s_soft[M][nn] = acc[v] + bias;
    }
  }
  __syncthreads();

  // ---------------- Stage 4a: softmax over K=64 (in place) -------------------
  {
    const int p = tid >> 4;
    const int j = (tid & 15) * 4;          // 4 cluster slots per lane
    float l0 = s_soft[p][j+0], l1 = s_soft[p][j+1];
    float l2 = s_soft[p][j+2], l3 = s_soft[p][j+3];
    float mx = fmaxf(fmaxf(l0, l1), fmaxf(l2, l3));
    #pragma unroll
    for (int m = 1; m < 16; m <<= 1) mx = fmaxf(mx, __shfl_xor(mx, m, 16));
    float e0 = __expf(l0 - mx), e1 = __expf(l1 - mx);
    float e2 = __expf(l2 - mx), e3 = __expf(l3 - mx);
    float sum = e0 + e1 + e2 + e3;
    #pragma unroll
    for (int m = 1; m < 16; m <<= 1) sum += __shfl_xor(sum, m, 16);
    const float isum = 1.0f / sum;
    s_soft[p][j+0] = e0 * isum;  s_soft[p][j+1] = e1 * isum;
    s_soft[p][j+2] = e2 * isum;  s_soft[p][j+3] = e3 * isum;
  }
  __syncthreads();

  // ---------------- Stage 4b: per-(pixel,k) residual norms -------------------
  for (int pk = tid; pk < TILE * KB; pk += 256) {    // 4 iterations
    const int p = pk >> 6;
    const int k = pk & 63;
    float ss = 0.0f;
    #pragma unroll 8
    for (int f = 0; f < FB; f += 4) {
      v4f xv = *(const v4f*)&s_xn[p][f];
      v4f cv = *(const v4f*)&s_cent[k][f];
      v4f d  = xv - cv;
      ss += d.x*d.x + d.y*d.y + d.z*d.z + d.w*d.w;
    }
    s_nk[p][k] = s_soft[p][k] * sqrtf(ss);   // ||(xn-cent)*soft||
  }
  __syncthreads();

  // ---------------- Stage 4c: final norm + per-(p,k) output scale ------------
  {
    const int p = tid >> 4;
    const int j = (tid & 15) * 4;
    float nk0 = s_nk[p][j+0], nk1 = s_nk[p][j+1];
    float nk2 = s_nk[p][j+2], nk3 = s_nk[p][j+3];
    float q0 = nk0 / fmaxf(nk0, EPSV), q1 = nk1 / fmaxf(nk1, EPSV);
    float q2 = nk2 / fmaxf(nk2, EPSV), q3 = nk3 / fmaxf(nk3, EPSV);
    float acc = q0*q0 + q1*q1 + q2*q2 + q3*q3;   // block-wise sumsq after norm
    #pragma unroll
    for (int m = 1; m < 16; m <<= 1) acc += __shfl_xor(acc, m, 16);
    const float invTot = 1.0f / fmaxf(sqrtf(acc), EPSV);
    s_scale[p][j+0] = s_soft[p][j+0] * (invTot / fmaxf(nk0, EPSV));
    s_scale[p][j+1] = s_soft[p][j+1] * (invTot / fmaxf(nk1, EPSV));
    s_scale[p][j+2] = s_soft[p][j+2] * (invTot / fmaxf(nk2, EPSV));
    s_scale[p][j+3] = s_soft[p][j+3] * (invTot / fmaxf(nk3, EPSV));
  }
  __syncthreads();

  // ---------------- Stage 5: stream 16 x 32KB output (non-temporal) ----------
  for (int p = 0; p < TILE; ++p) {
    float* op = out + (gp0 + p) * (long)(KB * FB);
    for (int i = tid; i < KB * FB / 4; i += 256) {   // 8 float4 stores / iter
      const int k = i >> 5;
      const int f = (i & 31) * 4;
      v4f xv = *(const v4f*)&s_xn[p][f];
      v4f cv = *(const v4f*)&s_cent[k][f];
      const float sc = s_scale[p][k];
      v4f o = (xv - cv) * sc;
      __builtin_nontemporal_store(o, (v4f*)(op + k * FB + f));
    }
  }
}

extern "C" void kernel_launch(void* const* d_in, const int* in_sizes, int n_in,
                              void* d_out, int out_size, void* d_ws, size_t ws_size,
                              hipStream_t stream) {
  (void)in_sizes; (void)n_in; (void)out_size; (void)d_ws; (void)ws_size;
  const float* x       = (const float*)d_in[0];
  const float* cluster = (const float*)d_in[1];
  const float* conv_w  = (const float*)d_in[2];
  const float* conv_b  = (const float*)d_in[3];
  const int*   row_idx = (const int*)d_in[4];
  float* out = (float*)d_out;

  dim3 grid(NB * TPN);   // 392 tiles of 16 pixels
  dim3 block(256);       // 8 wave32
  vlad_fused<<<grid, block, 0, stream>>>(x, cluster, conv_w, conv_b, row_idx, out);
}
